// Encoder_63866163691779
// MI455X (gfx1250) — compile-verified, matrix operand order checked by reading, not run on hardware
//
#include <hip/hip_runtime.h>

#define D_MODEL 512
#define D_FF    2048
#define SEQ     2048
#define BATCH   2
#define NROWS   (BATCH*SEQ)     // 4096
#define NEGINF  -1e9f

typedef __attribute__((ext_vector_type(16))) __bf16       v16bf;
typedef __attribute__((ext_vector_type(8)))  float        v8f;
typedef __attribute__((ext_vector_type(4)))  unsigned int v4u;

union FragBF { v16bf v; v4u q[2]; __bf16 e[16]; };
union Pack8  { v4u q; __bf16 e[8]; };

__device__ __forceinline__ __bf16 f2bf(float f) { return (__bf16)f; }

__device__ __forceinline__ v8f wmma_bf16(v16bf a, v16bf b, v8f c) {
  return __builtin_amdgcn_wmma_f32_16x16x32_bf16(false, a, false, b, (short)0, c,
                                                 false, false);
}

// Low 32 bits of a generic LDS pointer are the LDS byte offset (ISA aperture rule).
__device__ __forceinline__ unsigned lds_off(const void* p) {
  return (unsigned)(size_t)p;
}

// Issue one 16B async copy global -> LDS (tracked by ASYNCcnt).
__device__ __forceinline__ void async_cp16(unsigned lds_addr, const void* gaddr) {
  asm volatile("global_load_async_to_lds_b128 %0, %1, off"
               :: "v"(lds_addr), "v"(gaddr) : "memory");
}
__device__ __forceinline__ void wait_async0() {
  asm volatile("s_wait_asynccnt 0x0" ::: "memory");
}

// Load a 16x32 bf16 fragment (A-operand layout) from LDS.
// Lane l: row = l&15, half = l>>4; e[0..7]=K(half*8..+7), e[8..15]=K(16+half*8..+7)
__device__ __forceinline__ v16bf ld_frag(const __bf16* p0, int stride) {
  int lane = threadIdx.x & 31;
  int r = lane & 15, hf = lane >> 4;
  const __bf16* p = p0 + r * stride + hf * 8;
  FragBF f;
  f.q[0] = *(const v4u*)(p);
  f.q[1] = *(const v4u*)(p + 16);
  return f.v;
}

// ---------------------------------------------------------------------------
// GEMM: C[M,N] = A[M,K] * Bt[N,K]^T + bias[N].  A, Bt are bf16; both tiles are
// staged with async-to-LDS byte copies into double-buffered LDS.
// Block = 256 threads (8 waves, 4x2), block tile 128x128, wave tile 32x64.
// mode: bit0 = ReLU, bit1 = bf16 output.
// ---------------------------------------------------------------------------
#define BM  128
#define BNN 128
#define BK  32
#define LDT 40   // LDS tile row stride in bf16 (80B, 16B aligned)

__global__ __launch_bounds__(256) void gemm_bf16_kernel(
    const __bf16* __restrict__ A, const __bf16* __restrict__ Bt,
    const float* __restrict__ bias, void* __restrict__ Cout,
    int M, int N, int K, int mode)
{
  __shared__ __bf16 As[2][BM  * LDT];
  __shared__ __bf16 Bs[2][BNN * LDT];

  int tid = threadIdx.x;
  int lane = tid & 31, wid = tid >> 5;
  int r = lane & 15, hf = lane >> 4;
  int wm = (wid >> 1) * 32;          // 4 waves along M
  int wn = (wid & 1) * 64;           // 2 waves along N
  int m0 = blockIdx.y * BM, n0 = blockIdx.x * BNN;

  const size_t rowb = (size_t)K * 2;           // row bytes of A and Bt
  const char* Ab = (const char*)A  + (size_t)m0 * rowb;
  const char* Bb = (const char*)Bt + (size_t)n0 * rowb;

  // stage one 128-row x 64-byte tile (k-slice of 32 bf16) into LDS buffer
  auto stage = [&](int k0, int buf) {
    unsigned la = lds_off(&As[buf][0]);
    unsigned lb = lds_off(&Bs[buf][0]);
    #pragma unroll
    for (int i = 0; i < 2; ++i) {
      int lin = i * 256 + tid;        // 0..511
      int row = lin >> 2;
      int c   = (lin & 3) << 4;       // 0,16,32,48
      async_cp16(la + row * 80 + c, Ab + (size_t)row * rowb + (size_t)k0 * 2 + c);
      async_cp16(lb + row * 80 + c, Bb + (size_t)row * rowb + (size_t)k0 * 2 + c);
    }
  };

  v8f zero8 = {0.f,0.f,0.f,0.f,0.f,0.f,0.f,0.f};
  v8f acc[2][4];
  #pragma unroll
  for (int mt = 0; mt < 2; ++mt)
    #pragma unroll
    for (int nt = 0; nt < 4; ++nt) acc[mt][nt] = zero8;

  int buf = 0;
  stage(0, 0);
  for (int k0 = 0; k0 < K; k0 += BK) {
    wait_async0();
    __syncthreads();
    if (k0 + BK < K) stage(k0 + BK, buf ^ 1);

    v16bf af[2], bfr[4];
    #pragma unroll
    for (int mt = 0; mt < 2; ++mt)
      af[mt] = ld_frag(&As[buf][(wm + mt * 16) * LDT], LDT);
    #pragma unroll
    for (int nt = 0; nt < 4; ++nt)
      bfr[nt] = ld_frag(&Bs[buf][(wn + nt * 16) * LDT], LDT);
    #pragma unroll
    for (int mt = 0; mt < 2; ++mt)
      #pragma unroll
      for (int nt = 0; nt < 4; ++nt)
        acc[mt][nt] = wmma_bf16(af[mt], bfr[nt], acc[mt][nt]);
    buf ^= 1;
  }

  #pragma unroll
  for (int mt = 0; mt < 2; ++mt)
    #pragma unroll
    for (int nt = 0; nt < 4; ++nt)
      #pragma unroll
      for (int v = 0; v < 8; ++v) {
        int gm = m0 + wm + mt * 16 + v + 8 * hf;
        int gn = n0 + wn + nt * 16 + r;
        float val = acc[mt][nt][v] + bias[gn];
        if (mode & 1) val = fmaxf(val, 0.f);
        if (mode & 2) ((__bf16*)Cout)[(size_t)gm * N + gn] = f2bf(val);
        else          ((float*)Cout)[(size_t)gm * N + gn] = val;
      }
}

// ---------------------------------------------------------------------------
// Flash attention over bf16 Q/K/V, fp32 output.
// grid = (S/128, H, B), block = 256 (8 waves); wave owns 16 query rows.
// K tile staged with async-to-LDS (already in B-operand [key][dim] layout);
// V transposed via LDS scatter. Reference quirk: scaled score==0 -> -1e9.
// ---------------------------------------------------------------------------
__global__ __launch_bounds__(256) void attn_kernel(
    const __bf16* __restrict__ Q, const __bf16* __restrict__ Kp,
    const __bf16* __restrict__ Vp, float* __restrict__ Out)
{
  __shared__ __bf16 Ks[32 * 72];        // [key][dim], 144B rows
  __shared__ __bf16 Vs[64 * 40];        // [dim][key] (transposed)
  __shared__ __bf16 Ps[8 * 16 * 40];    // per-wave P scratch [qrow][key]

  int tid = threadIdx.x, lane = tid & 31, wid = tid >> 5;
  int r = lane & 15, hf = lane >> 4;
  int b = blockIdx.z, hh = blockIdx.y, qt = blockIdx.x;
  int qrow0 = b * SEQ + qt * 128 + wid * 16;
  unsigned ksb = lds_off(Ks);

  // Q fragments (16 rows x 64 dims -> two 16x32 A-fragments), direct bf16 loads
  FragBF qa[2];
  const char* qp = (const char*)Q + ((size_t)(qrow0 + r) * D_MODEL + hh * 64) * 2;
  #pragma unroll
  for (int kb = 0; kb < 2; ++kb)
    #pragma unroll
    for (int g = 0; g < 2; ++g)
      qa[kb].q[g] = *(const v4u*)(qp + kb * 64 + g * 32 + hf * 16);

  v8f zero8 = {0.f,0.f,0.f,0.f,0.f,0.f,0.f,0.f};
  v8f O[4]; O[0] = zero8; O[1] = zero8; O[2] = zero8; O[3] = zero8;
  float m_run[8], l_run[8];
  #pragma unroll
  for (int v = 0; v < 8; ++v) { m_run[v] = -1e30f; l_run[v] = 0.f; }

  __bf16* Pw = &Ps[wid * 16 * 40];

  for (int kt = 0; kt < SEQ; kt += 32) {
    __syncthreads();
    {
      int key = tid >> 3;
      size_t grow = ((size_t)(b * SEQ + kt + key) * D_MODEL + hh * 64) * 2;
      // K tile: async byte copy, one 16B chunk per thread (32 keys x 128B)
      int c = (tid & 7) << 4;
      async_cp16(ksb + key * 144 + c, (const char*)Kp + grow + c);
      // V tile: load 8 bf16, scatter transposed
      int d8 = (tid & 7) << 3;
      Pack8 vv;
      vv.q = *(const v4u*)((const char*)Vp + grow + d8 * 2);
      #pragma unroll
      for (int j = 0; j < 8; ++j) Vs[(d8 + j) * 40 + key] = vv.e[j];
    }
    wait_async0();
    __syncthreads();

    // Scores: two 16x16 tiles over the 32-key window (contraction 64 = 2 steps)
    v8f s0 = zero8, s1 = zero8;
    v16bf k00 = ld_frag(&Ks[0], 72);
    v16bf k01 = ld_frag(&Ks[32], 72);
    v16bf k10 = ld_frag(&Ks[16 * 72], 72);
    v16bf k11 = ld_frag(&Ks[16 * 72 + 32], 72);
    s0 = wmma_bf16(qa[0].v, k00, s0);
    s0 = wmma_bf16(qa[1].v, k01, s0);
    s1 = wmma_bf16(qa[0].v, k10, s1);
    s1 = wmma_bf16(qa[1].v, k11, s1);

    float p0[8], p1[8];
    #pragma unroll
    for (int v = 0; v < 8; ++v) {
      float x0 = s0[v] * 0.125f; if (x0 == 0.f) x0 = NEGINF;  // reference quirk
      float x1 = s1[v] * 0.125f; if (x1 == 0.f) x1 = NEGINF;
      float mx = fmaxf(x0, x1);
      #pragma unroll
      for (int off = 8; off; off >>= 1) mx = fmaxf(mx, __shfl_xor(mx, off, 32));
      float mnew = fmaxf(m_run[v], mx);
      float e0 = __expf(x0 - mnew), e1 = __expf(x1 - mnew);
      float srow = e0 + e1;
      #pragma unroll
      for (int off = 8; off; off >>= 1) srow += __shfl_xor(srow, off, 32);
      float alpha = __expf(m_run[v] - mnew);
      l_run[v] = l_run[v] * alpha + srow;
      m_run[v] = mnew;
      #pragma unroll
      for (int j = 0; j < 4; ++j) O[j][v] *= alpha;
      p0[v] = e0; p1[v] = e1;
    }

    // Route P (C-layout) through per-wave LDS into A-fragment layout
    #pragma unroll
    for (int v = 0; v < 8; ++v) {
      Pw[(v + 8 * hf) * 40 + r]      = f2bf(p0[v]);
      Pw[(v + 8 * hf) * 40 + 16 + r] = f2bf(p1[v]);
    }
    asm volatile("s_wait_dscnt 0x0" ::: "memory");
    v16bf pf = ld_frag(Pw, 40);
    #pragma unroll
    for (int j = 0; j < 4; ++j) {
      v16bf vf = ld_frag(&Vs[j * 16 * 40], 40);
      O[j] = wmma_bf16(pf, vf, O[j]);
    }

    if (kt + 32 < SEQ)
      __builtin_prefetch((const char*)Kp +
          ((size_t)(b * SEQ + kt + 32 + r) * D_MODEL + hh * 64) * 2 + hf * 64, 0, 1);
  }

  #pragma unroll
  for (int j = 0; j < 4; ++j)
    #pragma unroll
    for (int v = 0; v < 8; ++v) {
      int row = qrow0 + v + 8 * hf;
      int col = hh * 64 + j * 16 + r;
      Out[(size_t)row * D_MODEL + col] = O[j][v] / l_run[v];
    }
}

// ---------------------------------------------------------------------------
// out = LayerNorm(a + b) * g + beta; optional bf16 mirror for next GEMM.
// ---------------------------------------------------------------------------
__global__ __launch_bounds__(256) void add_norm_kernel(
    const float* __restrict__ A, const float* __restrict__ Bv,
    const float* __restrict__ g, const float* __restrict__ be,
    float* __restrict__ out, __bf16* __restrict__ outbf)
{
  int row = blockIdx.x, t = threadIdx.x;
  int lane = t & 31, wid = t >> 5;
  const float* a = A + (size_t)row * D_MODEL;
  const float* b = Bv + (size_t)row * D_MODEL;
  float x0 = a[t] + b[t];
  float x1 = a[t + 256] + b[t + 256];

  __shared__ float sred[8];
  __shared__ float stat;

  float s = x0 + x1;
  #pragma unroll
  for (int off = 16; off; off >>= 1) s += __shfl_xor(s, off, 32);
  if (lane == 0) sred[wid] = s;
  __syncthreads();
  if (t == 0) {
    float tot = 0.f;
    for (int i = 0; i < 8; ++i) tot += sred[i];
    stat = tot * (1.0f / D_MODEL);
  }
  __syncthreads();
  float mean = stat;
  float d0 = x0 - mean, d1 = x1 - mean;
  float q = d0 * d0 + d1 * d1;
  #pragma unroll
  for (int off = 16; off; off >>= 1) q += __shfl_xor(q, off, 32);
  __syncthreads();
  if (lane == 0) sred[wid] = q;
  __syncthreads();
  if (t == 0) {
    float tot = 0.f;
    for (int i = 0; i < 8; ++i) tot += sred[i];
    stat = tot * (1.0f / D_MODEL);
  }
  __syncthreads();
  float inv = rsqrtf(stat + 1e-5f);
  float r0 = d0 * inv * g[t] + be[t];
  float r1 = d1 * inv * g[t + 256] + be[t + 256];
  out[(size_t)row * D_MODEL + t]       = r0;
  out[(size_t)row * D_MODEL + t + 256] = r1;
  if (outbf) {
    outbf[(size_t)row * D_MODEL + t]       = f2bf(r0);
    outbf[(size_t)row * D_MODEL + t + 256] = f2bf(r1);
  }
}

// ---------------------------------------------------------------------------
// Wt[n][k] = bf16(W[k][n]) — tiled transpose + convert (runs once per launch).
// block (32,8), grid (N/32, K/32).
// ---------------------------------------------------------------------------
__global__ void transpose_convert_kernel(const float* __restrict__ in,
                                         __bf16* __restrict__ out, int K, int N)
{
  __shared__ float tl[32][33];
  int bx = blockIdx.x * 32;   // N
  int by = blockIdx.y * 32;   // K
  int tx = threadIdx.x;
  for (int i = threadIdx.y; i < 32; i += 8)
    tl[i][tx] = in[(size_t)(by + i) * N + bx + tx];
  __syncthreads();
  for (int i = threadIdx.y; i < 32; i += 8)
    out[(size_t)(bx + i) * K + by + tx] = f2bf(tl[tx][i]);
}

__global__ void f32_to_bf16_kernel(const float* __restrict__ in,
                                   __bf16* __restrict__ out)
{
  size_t i = (size_t)blockIdx.x * 256 + threadIdx.x;
  out[i] = f2bf(in[i]);
}

// ---------------------------------------------------------------------------
extern "C" void kernel_launch(void* const* d_in, const int* in_sizes, int n_in,
                              void* d_out, int out_size, void* d_ws, size_t ws_size,
                              hipStream_t stream) {
  (void)in_sizes; (void)n_in; (void)out_size; (void)ws_size;
  const float* x     = (const float*)d_in[0];
  const float* Wq    = (const float*)d_in[1];
  const float* bq    = (const float*)d_in[2];
  const float* Wk    = (const float*)d_in[3];
  const float* bk    = (const float*)d_in[4];
  const float* Wv    = (const float*)d_in[5];
  const float* bv    = (const float*)d_in[6];
  const float* W1    = (const float*)d_in[7];
  const float* b1    = (const float*)d_in[8];
  const float* W2    = (const float*)d_in[9];
  const float* b2    = (const float*)d_in[10];
  const float* g1    = (const float*)d_in[11];
  const float* beta1 = (const float*)d_in[12];
  const float* g2    = (const float*)d_in[13];
  const float* beta2 = (const float*)d_in[14];

  const size_t SDe = (size_t)NROWS * D_MODEL;   // 4096*512 elements
  char* p = (char*)d_ws;
  auto carve = [&](size_t bytes) -> void* {
    void* r = (void*)p; p += (bytes + 255) & ~(size_t)255; return r;
  };
  float*  atb   = (float*)carve(SDe * 4);
  float*  ln1   = (float*)carve(SDe * 4);
  float*  ff2   = (float*)carve(SDe * 4);
  float*  hb    = (float*)carve(SDe * 4);
  __bf16* qb    = (__bf16*)carve(SDe * 2);
  __bf16* kbb   = (__bf16*)carve(SDe * 2);
  __bf16* vbb   = (__bf16*)carve(SDe * 2);
  __bf16* hbf   = (__bf16*)carve(SDe * 2);
  __bf16* ln1bf = (__bf16*)carve(SDe * 2);
  __bf16* ff1bf = (__bf16*)carve((size_t)NROWS * D_FF * 2);
  __bf16* Wqt   = (__bf16*)carve((size_t)4 * D_MODEL * D_MODEL * 2);
  __bf16* Wkt   = (__bf16*)carve((size_t)4 * D_MODEL * D_MODEL * 2);
  __bf16* Wvt   = (__bf16*)carve((size_t)4 * D_MODEL * D_MODEL * 2);
  __bf16* W1t   = (__bf16*)carve((size_t)4 * D_MODEL * D_FF * 2);
  __bf16* W2t   = (__bf16*)carve((size_t)4 * D_FF * D_MODEL * 2);

  dim3 blk(256);
  dim3 tb(32, 8);
  const size_t WDD = (size_t)D_MODEL * D_MODEL;     // 262144
  const size_t WDF = (size_t)D_MODEL * D_FF;        // 1048576

  // One-time (per launch) weight convert+transpose to bf16 [N][K]
  for (int l = 0; l < 4; ++l) {
    transpose_convert_kernel<<<dim3(16, 16), tb, 0, stream>>>(Wq + l * WDD, Wqt + l * WDD, D_MODEL, D_MODEL);
    transpose_convert_kernel<<<dim3(16, 16), tb, 0, stream>>>(Wk + l * WDD, Wkt + l * WDD, D_MODEL, D_MODEL);
    transpose_convert_kernel<<<dim3(16, 16), tb, 0, stream>>>(Wv + l * WDD, Wvt + l * WDD, D_MODEL, D_MODEL);
    transpose_convert_kernel<<<dim3(64, 16), tb, 0, stream>>>(W1 + l * WDF, W1t + l * WDF, D_MODEL, D_FF);
    transpose_convert_kernel<<<dim3(16, 64), tb, 0, stream>>>(W2 + l * WDF, W2t + l * WDF, D_FF, D_MODEL);
  }

  hipMemcpyAsync(hb, x, SDe * sizeof(float), hipMemcpyDeviceToDevice, stream);
  f32_to_bf16_kernel<<<(int)(SDe / 256), blk, 0, stream>>>(x, hbf);

  dim3 gD (D_MODEL / BNN, NROWS / BM);   // (4, 32)
  dim3 gF1(D_FF    / BNN, NROWS / BM);   // (16, 32)
  dim3 gAttn(SEQ / 128, 8, BATCH);       // (16, 8, 2)

  for (int l = 0; l < 4; ++l) {
    const float* bql = bq + (size_t)l * D_MODEL;
    const float* bkl = bk + (size_t)l * D_MODEL;
    const float* bvl = bv + (size_t)l * D_MODEL;
    const float* b1l = b1 + (size_t)l * D_FF;
    const float* b2l = b2 + (size_t)l * D_MODEL;
    const float* g1l = g1 + (size_t)l * D_MODEL;
    const float* e1l = beta1 + (size_t)l * D_MODEL;
    const float* g2l = g2 + (size_t)l * D_MODEL;
    const float* e2l = beta2 + (size_t)l * D_MODEL;

    gemm_bf16_kernel<<<gD, blk, 0, stream>>>(hbf, Wqt + l * WDD, bql, qb,  NROWS, D_MODEL, D_MODEL, 2);
    gemm_bf16_kernel<<<gD, blk, 0, stream>>>(hbf, Wkt + l * WDD, bkl, kbb, NROWS, D_MODEL, D_MODEL, 2);
    gemm_bf16_kernel<<<gD, blk, 0, stream>>>(hbf, Wvt + l * WDD, bvl, vbb, NROWS, D_MODEL, D_MODEL, 2);
    attn_kernel<<<gAttn, blk, 0, stream>>>(qb, kbb, vbb, atb);
    add_norm_kernel<<<NROWS, blk, 0, stream>>>(atb, hb, g1l, e1l, ln1, ln1bf);
    gemm_bf16_kernel<<<gF1, blk, 0, stream>>>(ln1bf, W1t + l * WDF, b1l, ff1bf, NROWS, D_FF, D_MODEL, 3);
    gemm_bf16_kernel<<<gD,  blk, 0, stream>>>(ff1bf, W2t + l * WDF, b2l, ff2, NROWS, D_MODEL, D_FF, 0);
    float*  outp  = (l == 3) ? (float*)d_out : hb;
    __bf16* outbf = (l == 3) ? (__bf16*)nullptr : hbf;
    add_norm_kernel<<<NROWS, blk, 0, stream>>>(ln1, ff2, g2l, e2l, outp, outbf);
  }
}
